// MaxSim_73358041416037
// MI455X (gfx1250) — matrix operand order repaired and checked by reading, not compiled, not run on hardware
//
#include <hip/hip_runtime.h>
#include <math.h>

typedef __attribute__((ext_vector_type(16))) __bf16 v16bf;
typedef __attribute__((ext_vector_type(8)))  float  v8f;

#define DD     1024
#define BB     128
#define LQ     256
#define LK     512
#define NQROWS (BB * LQ)            // 32768 query rows
#define NKROWS (BB * LK)            // 65536 key rows per set
#define NROWS  (NQROWS + 2 * NKROWS)
#define QG     64                   // q rows per workgroup
#define NQG    (LQ / QG)            // 4
#define TPB    128                  // 4 waves (wave32)
#define NWAVE  4
#define NKT    (LK / 16)            // 32 key tiles

// ---------------- Pass 1: per-row 1/max(||x||, eps) ----------------
__global__ void rownorm_kernel(const float* __restrict__ q,
                               const float* __restrict__ pk,
                               const float* __restrict__ nk,
                               float* __restrict__ scales) {
  int r = blockIdx.x;
  const float* src;
  if (r < NQROWS)               src = q  + (size_t)r * DD;
  else if (r < NQROWS + NKROWS) src = pk + (size_t)(r - NQROWS) * DD;
  else                          src = nk + (size_t)(r - NQROWS - NKROWS) * DD;

  int tid = threadIdx.x;                       // 256 threads, 4 floats each
  float4 f = *(const float4*)(src + tid * 4);
  float ss = f.x * f.x + f.y * f.y + f.z * f.z + f.w * f.w;
  for (int off = 16; off >= 1; off >>= 1) ss += __shfl_xor(ss, off, 32);
  __shared__ float sb[8];
  if ((tid & 31) == 0) sb[tid >> 5] = ss;
  __syncthreads();
  if (tid == 0) {
    float t = 0.f;
    for (int i = 0; i < 8; ++i) t += sb[i];
    scales[r] = 1.0f / fmaxf(sqrtf(t), 1e-12f);
  }
}

// ---------------- Pass 2: bf16 WMMA maxsim ----------------
__global__ void __launch_bounds__(TPB) maxsim_wmma_kernel(
    const float* __restrict__ query, const float* __restrict__ pos_key,
    const float* __restrict__ neg_key, const int* __restrict__ query_mask,
    const int* __restrict__ pos_mask, const int* __restrict__ neg_mask,
    const float* __restrict__ scales, float* __restrict__ partials) {
  extern __shared__ __bf16 lds[];
  __bf16* qlds = lds;            // QG x DD bf16 (128 KB)
  __bf16* klds = lds + QG * DD;  // 16 x DD bf16 (32 KB)

  int wg   = blockIdx.x;         // wg = b*8 + set*4 + qg
  int qg   = wg & (NQG - 1);
  int set  = (wg >> 2) & 1;
  int b    = wg >> 3;
  int tid  = threadIdx.x;
  int lane = tid & 31;
  int wave = tid >> 5;
  int m    = lane & 15;          // row (A) / column (B) within tile
  int half = lane >> 4;

  const float* qptr   = query + ((size_t)b * LQ + qg * QG) * DD;
  const float* kptr   = (set ? neg_key : pos_key) + (size_t)b * LK * DD;
  const int*   kmask  = (set ? neg_mask : pos_mask) + b * LK;
  const int*   qmask  = query_mask + b * LQ + qg * QG;
  const float* qscale = scales + (size_t)b * LQ + qg * QG;
  const float* kscale = scales + NQROWS + (size_t)set * NKROWS + (size_t)b * LK;

  // Stage normalized q tile (QG x DD) as bf16 in LDS
  for (int idx = tid * 4; idx < QG * DD; idx += TPB * 4) {
    int row = idx >> 10;
    float s = qscale[row];
    float4 f = *(const float4*)(qptr + idx);
    qlds[idx + 0] = (__bf16)(f.x * s);
    qlds[idx + 1] = (__bf16)(f.y * s);
    qlds[idx + 2] = (__bf16)(f.z * s);
    qlds[idx + 3] = (__bf16)(f.w * s);
  }

  float rm[8];
  #pragma unroll
  for (int i = 0; i < 8; ++i) rm[i] = -3.402823466e38f;

  const __bf16* qrow = qlds + (size_t)(wave * 16 + m) * DD;  // A: row m of this wave's tile
  const __bf16* krow = klds + (size_t)m * DD;                // B: column m (key row m)

  for (int kt = 0; kt < NKT; ++kt) {
    __syncthreads();  // protect klds reads from previous iteration
    // Stage normalized 16-key tile as bf16
    for (int idx = tid * 4; idx < 16 * DD; idx += TPB * 4) {
      int row = idx >> 10;
      float s = kscale[kt * 16 + row];
      float4 f = *(const float4*)(kptr + (size_t)kt * 16 * DD + idx);
      klds[idx + 0] = (__bf16)(f.x * s);
      klds[idx + 1] = (__bf16)(f.y * s);
      klds[idx + 2] = (__bf16)(f.z * s);
      klds[idx + 3] = (__bf16)(f.w * s);
    }
    if (kt + 1 < NKT)  // global_prefetch_b8 the next key tile
      __builtin_prefetch((const void*)(kptr + (size_t)(kt + 1) * 16 * DD + tid * 128), 0, 0);
    __syncthreads();

    v8f acc = {0.f, 0.f, 0.f, 0.f, 0.f, 0.f, 0.f, 0.f};
    #pragma unroll 4
    for (int dc = 0; dc < DD; dc += 32) {
      union { uint4 u[2]; v16bf v; } ua, ub;
      // A lane layout: half 0 -> K {0..7,16..23}; half 1 -> K {8..15,24..31}
      ua.u[0] = *(const uint4*)(qrow + dc + 8 * half);
      ua.u[1] = *(const uint4*)(qrow + dc + 16 + 8 * half);
      // B lane layout: half 0 -> K {0..7,16..23}; half 1 -> K {8..15,24..31}
      ub.u[0] = *(const uint4*)(krow + dc + 8 * half);
      ub.u[1] = *(const uint4*)(krow + dc + 16 + 8 * half);
      acc = __builtin_amdgcn_wmma_f32_16x16x32_bf16(
          false, ua.v, false, ub.v, (short)0, acc, false, false);
    }
    // C/D layout: acc[i], lanes 0-15 -> (row i, col m); lanes 16-31 -> (row i+8, col m)
    float cm = (float)kmask[kt * 16 + m];  // multiplicative key mask per column
    #pragma unroll
    for (int i = 0; i < 8; ++i) rm[i] = fmaxf(rm[i], acc[i] * cm);
  }

  // Max over the 16 columns held across each 16-lane half
  #pragma unroll
  for (int i = 0; i < 8; ++i)
    for (int off = 8; off >= 1; off >>= 1)
      rm[i] = fmaxf(rm[i], __shfl_xor(rm[i], off, 16));

  if (m == 0) {  // lanes 0 (rows i) and 16 (rows i+8)
    float part = 0.f;
    int rbase = wave * 16 + half * 8;
    #pragma unroll
    for (int i = 0; i < 8; ++i) part += rm[i] * (float)qmask[rbase + i];
    part += __shfl_xor(part, 16, 32);  // combine the two halves
    if (lane == 0)
      partials[((size_t)(b * 2 + set) * NQG + qg) * NWAVE + wave] = part;
  }
}

// ---------------- Pass 3: logits -> CE -> mean ----------------
__global__ void finalize_kernel(const float* __restrict__ partials,
                                float* __restrict__ out) {
  __shared__ float red[BB];
  int b = threadIdx.x;
  const float* pp = partials + (size_t)(b * 2) * (NQG * NWAVE);
  float pos = 0.f, neg = 0.f;
  for (int j = 0; j < NQG * NWAVE; ++j) {
    pos += pp[j];
    neg += pp[NQG * NWAVE + j];
  }
  float lp = pos * 20.0f;  // 1 / TEMPERATURE
  float ln = neg * 20.0f;
  float mx = fmaxf(lp, ln);
  float nll = (mx - lp) + logf(expf(lp - mx) + expf(ln - mx));
  red[b] = nll;
  __syncthreads();
  if (b == 0) {
    float t = 0.f;
    for (int i = 0; i < BB; ++i) t += red[i];
    out[0] = t / (float)BB;
  }
}

extern "C" void kernel_launch(void* const* d_in, const int* in_sizes, int n_in,
                              void* d_out, int out_size, void* d_ws, size_t ws_size,
                              hipStream_t stream) {
  const float* query      = (const float*)d_in[0];
  const float* pos_key    = (const float*)d_in[1];
  const float* neg_key    = (const float*)d_in[2];
  const int*   query_mask = (const int*)d_in[3];
  const int*   pos_mask   = (const int*)d_in[4];
  const int*   neg_mask   = (const int*)d_in[5];

  float* ws       = (float*)d_ws;
  float* scales   = ws;            // NROWS floats (640 KB)
  float* partials = ws + NROWS;    // 128*2*16 floats

  rownorm_kernel<<<NROWS, 256, 0, stream>>>(query, pos_key, neg_key, scales);

  size_t lds_bytes = (size_t)(QG * DD + 16 * DD) * sizeof(__bf16);  // 160 KB (< 320 KB/WGP)
  maxsim_wmma_kernel<<<BB * 2 * NQG, TPB, lds_bytes, stream>>>(
      query, pos_key, neg_key, query_mask, pos_mask, neg_mask, scales, partials);

  finalize_kernel<<<1, BB, 0, stream>>>(partials, (float*)d_out);
}
